// EnhancedGNNPolicyValueNetwork_38946763440393
// MI455X (gfx1250) — compile-verified, matrix operand order checked by reading, not run on hardware
//
#include <hip/hip_runtime.h>
#include <hip/hip_bf16.h>

#define NN 50000
#define NE 800000
#define FNODE 120
#define FEDGE 16
#define HIDC 128
#define NHEADS 4
#define HEDIM 64

typedef __attribute__((ext_vector_type(16))) _Float16 v16h;
typedef __attribute__((ext_vector_type(8)))  float    v8f;

__device__ __forceinline__ float lrelu02(float x) { return x > 0.f ? x : 0.2f * x; }

__device__ __forceinline__ void atomicMaxFloat(float* addr, float val) {
    if (val >= 0.f) atomicMax((int*)addr, __float_as_int(val));
    else            atomicMin((unsigned int*)addr, __float_as_uint(val));
}

// 16x32 f16 WMMA fragment (CDNA5 ISA 7.12.2 A-operand layout) from fp32
// row-major [rows, K]. Lane L holds row (L&15); lanes 0-15 carry K offsets
// {k0..k0+7, k0+16..k0+23}, lanes 16-31 carry {k0+8..k0+15, k0+24..k0+31}.
// K is compile-time: the only guard is the uniform, foldable k0+32<=K
// (half-fragment granularity), so loads are 4x b128 with no branches.
template<int K>
__device__ __forceinline__ v16h frag16x32(const float* __restrict__ base,
                                          int row0, int k0, int lane) {
    int row = row0 + (lane & 15);
    int kb  = k0 + ((lane & 16) ? 8 : 0);
    const float* p = base + (size_t)row * K + kb;
    float4 a0 = *(const float4*)(p);
    float4 a1 = *(const float4*)(p + 4);
    float4 b0, b1;
    if (k0 + 32 <= K) {                 // compile-time after unroll
        b0 = *(const float4*)(p + 16);
        b1 = *(const float4*)(p + 20);
    } else {
        b0 = make_float4(0.f, 0.f, 0.f, 0.f);
        b1 = b0;
    }
    v16h f;
    f[0]  = (_Float16)a0.x; f[1]  = (_Float16)a0.y;
    f[2]  = (_Float16)a0.z; f[3]  = (_Float16)a0.w;
    f[4]  = (_Float16)a1.x; f[5]  = (_Float16)a1.y;
    f[6]  = (_Float16)a1.z; f[7]  = (_Float16)a1.w;
    f[8]  = (_Float16)b0.x; f[9]  = (_Float16)b0.y;
    f[10] = (_Float16)b0.z; f[11] = (_Float16)b0.w;
    f[12] = (_Float16)b1.x; f[13] = (_Float16)b1.y;
    f[14] = (_Float16)b1.z; f[15] = (_Float16)b1.w;
    return f;
}

// C[M,Nout] = act(A[M,K] @ W[Nout,K]^T + bias).
// 1x4 register blocking: each wave32 computes a 16x64 strip (4 accumulators),
// reusing one A fragment across 4 WMMAs -> 5 b128 loads per WMMA instead of 8.
// Requires M % 16 == 0 and Nout % 64 == 0 (true for all uses here).
template<int K>
__global__ void wmma_gemm(const float* __restrict__ A, const float* __restrict__ W,
                          const float* __restrict__ bias, float* __restrict__ C,
                          int M, int Nout, int do_relu) {
    int wid   = blockIdx.x * (blockDim.x >> 5) + (threadIdx.x >> 5);
    int lane  = threadIdx.x & 31;
    int ntile = Nout >> 6;              // 64-wide strips
    int total = (M >> 4) * ntile;
    if (wid >= total) return;           // wave-uniform: EXEC all ones at WMMA
    int tm = wid / ntile, tn = wid % ntile;
    v8f acc0 = {}, acc1 = {}, acc2 = {}, acc3 = {};
#pragma unroll
    for (int k0 = 0; k0 < K; k0 += 32) {
        __builtin_prefetch(A + (size_t)(tm * 16 + (lane & 15)) * K + k0 + 64, 0, 1);
        v16h a  = frag16x32<K>(A, tm * 16, k0, lane);
        v16h b0 = frag16x32<K>(W, tn * 64,      k0, lane);
        v16h b1 = frag16x32<K>(W, tn * 64 + 16, k0, lane);
        v16h b2 = frag16x32<K>(W, tn * 64 + 32, k0, lane);
        v16h b3 = frag16x32<K>(W, tn * 64 + 48, k0, lane);
        acc0 = __builtin_amdgcn_wmma_f32_16x16x32_f16(false, a, false, b0, (short)0, acc0, false, false);
        acc1 = __builtin_amdgcn_wmma_f32_16x16x32_f16(false, a, false, b1, (short)0, acc1, false, false);
        acc2 = __builtin_amdgcn_wmma_f32_16x16x32_f16(false, a, false, b2, (short)0, acc2, false, false);
        acc3 = __builtin_amdgcn_wmma_f32_16x16x32_f16(false, a, false, b3, (short)0, acc3, false, false);
    }
    // C/D layout: VGPR i -> row m0+i; lanes 0-15 N=lane (M 0-7),
    // lanes 16-31 N=lane-16 (M 8-15).
    int m0 = tm * 16 + ((lane & 16) ? 8 : 0);
    v8f accs[4] = {acc0, acc1, acc2, acc3};
#pragma unroll
    for (int t = 0; t < 4; ++t) {
        int n  = tn * 64 + t * 16 + (lane & 15);
        float bv = bias ? bias[n] : 0.f;
#pragma unroll
        for (int i = 0; i < 8; ++i) {
            float v = accs[t][i] + bv;
            if (do_relu) v = fmaxf(v, 0.f);
            C[(size_t)(m0 + i) * Nout + n] = v;
        }
    }
}

__global__ void k_zero(float* p, int n) {
    int i = blockIdx.x * blockDim.x + threadIdx.x;
    if (i < n) p[i] = 0.f;
}

__global__ void k_concat(const float* __restrict__ x, const int* __restrict__ nt,
                         const float* __restrict__ emb, float* __restrict__ hin) {
    int i = blockIdx.x * blockDim.x + threadIdx.x;
    if (i >= NN * 128) return;
    int n = i >> 7, c = i & 127;
    hin[i] = (c < FNODE) ? x[(size_t)n * FNODE + c] : emb[nt[n] * 8 + (c - FNODE)];
}

__global__ void k_deg_esum(const float* __restrict__ e, const int* __restrict__ ei,
                           float* __restrict__ esum, float* __restrict__ deg) {
    int i = blockIdx.x * blockDim.x + threadIdx.x;
    if (i >= NE * 64) return;
    int ed = i >> 6, c = i & 63;
    int d = ei[NE + ed];
    atomicAdd(&esum[(size_t)d * 64 + c], e[i]);
    if (c == 0) atomicAdd(&deg[d], 1.f);
}

__global__ void k_loopattr(float* __restrict__ esum, const float* __restrict__ deg) {
    int i = blockIdx.x * blockDim.x + threadIdx.x;
    if (i >= NN * 64) return;
    esum[i] = esum[i] / fmaxf(deg[i >> 6], 1.f);
}

__device__ __forceinline__ float dot4n(const float4* __restrict__ a,
                                       const float4* __restrict__ b, int n4) {
    float s = 0.f;
#pragma unroll 8
    for (int k = 0; k < n4; ++k) {
        float4 u = a[k], v = b[k];
        s += u.x * v.x + u.y * v.y + u.z * v.z + u.w * v.w;
    }
    return s;
}

// a_s[n,h] = xp[n,h,:].a_src[h,:] ; a_d likewise
__global__ void k_attdot(const float* __restrict__ xp, const float* __restrict__ asrc,
                         const float* __restrict__ adst, float* __restrict__ as_,
                         float* __restrict__ ad_, int H, int C) {
    int i = blockIdx.x * blockDim.x + threadIdx.x;
    if (i >= NN * H) return;
    int n = i / H, h = i % H;
    const float4* xr = (const float4*)(xp + (size_t)n * H * C + (size_t)h * C);
    as_[i] = dot4n(xr, (const float4*)(asrc + h * C), C / 4);
    ad_[i] = dot4n(xr, (const float4*)(adst + h * C), C / 4);
}

// M[h,k] = sum_c We[h*C+c, k] * ae[h, c]   (folds a_edge into edge projection)
__global__ void k_foldW(const float* __restrict__ We, const float* __restrict__ ae,
                        float* __restrict__ M, int H, int C) {
    int i = blockIdx.x * blockDim.x + threadIdx.x;
    if (i >= H * HEDIM) return;
    int h = i / HEDIM, k = i % HEDIM;
    float s = 0.f;
    for (int c = 0; c < C; ++c) s += We[((size_t)h * C + c) * HEDIM + k] * ae[h * C + c];
    M[i] = s;
}

__global__ void k_selfloop(const float* __restrict__ la, const float* __restrict__ M,
                           const float* __restrict__ as_, const float* __restrict__ ad_,
                           float* __restrict__ lsl, float* __restrict__ m, int H) {
    int i = blockIdx.x * blockDim.x + threadIdx.x;
    if (i >= NN * H) return;
    int n = i / H, h = i % H;
    float ae = dot4n((const float4*)(la + (size_t)n * HEDIM),
                     (const float4*)(M + h * HEDIM), HEDIM / 4);
    float l = lrelu02(as_[i] + ad_[i] + ae);
    lsl[i] = l; m[i] = l;
}

__global__ void k_edgelogit(const float* __restrict__ e, const int* __restrict__ ei,
                            const float* __restrict__ M, const float* __restrict__ as_,
                            const float* __restrict__ ad_, float* __restrict__ logits,
                            float* __restrict__ m, int H) {
    int i = blockIdx.x * blockDim.x + threadIdx.x;
    if (i >= NE * H) return;
    int ed = i / H, h = i % H;
    int s = ei[ed], d = ei[NE + ed];
    float ae = dot4n((const float4*)(e + (size_t)ed * HEDIM),
                     (const float4*)(M + h * HEDIM), HEDIM / 4);
    float l = lrelu02(as_[s * H + h] + ad_[d * H + h] + ae);
    logits[i] = l;
    atomicMaxFloat(&m[d * H + h], l);
}

__global__ void k_denom_init(const float* __restrict__ lsl, const float* __restrict__ m,
                             float* __restrict__ dn, int H) {
    int i = blockIdx.x * blockDim.x + threadIdx.x;
    if (i >= NN * H) return;
    dn[i] = __expf(lsl[i] - m[i]);
}

__global__ void k_denom_edge(const float* __restrict__ logits, const int* __restrict__ ei,
                             const float* __restrict__ m, float* __restrict__ dn, int H) {
    int i = blockIdx.x * blockDim.x + threadIdx.x;
    if (i >= NE * H) return;
    int ed = i / H, h = i % H;
    int d = ei[NE + ed];
    atomicAdd(&dn[d * H + h], __expf(logits[i] - m[d * H + h]));
}

// out[n, :] = alpha_self(n,h) * xp[n, :]   (float4 chunks)
__global__ void k_agg_init(const float* __restrict__ xp, const float* __restrict__ lsl,
                           const float* __restrict__ m, const float* __restrict__ dn,
                           float* __restrict__ out, int H, int C) {
    int q4 = H * C / 4;
    int i = blockIdx.x * blockDim.x + threadIdx.x;
    if (i >= NN * q4) return;
    int n = i / q4, off = (i % q4) * 4;
    int h = off / C;
    float a = __expf(lsl[n * H + h] - m[n * H + h]) / dn[n * H + h];
    float4 v = *(const float4*)(xp + (size_t)n * H * C + off);
    float4 r; r.x = a * v.x; r.y = a * v.y; r.z = a * v.z; r.w = a * v.w;
    *(float4*)(out + (size_t)n * H * C + off) = r;
}

__global__ void k_agg_edge(const float* __restrict__ xp, const int* __restrict__ ei,
                           const float* __restrict__ logits, const float* __restrict__ m,
                           const float* __restrict__ dn, float* __restrict__ out, int H, int C) {
    int q4 = H * C / 4;
    int i = blockIdx.x * blockDim.x + threadIdx.x;
    if (i >= NE * q4) return;
    int ed = i / q4, off = (i % q4) * 4;
    int h = off / C;
    int s = ei[ed], d = ei[NE + ed];
    float a = __expf(logits[(size_t)ed * H + h] - m[d * H + h]) / dn[d * H + h];
    float4 v = *(const float4*)(xp + (size_t)s * H * C + off);
    float* o = out + (size_t)d * H * C + off;
    atomicAdd(o + 0, a * v.x); atomicAdd(o + 1, a * v.y);
    atomicAdd(o + 2, a * v.z); atomicAdd(o + 3, a * v.w);
}

__global__ void k_bias_relu(float* __restrict__ h, const float* __restrict__ b, int D) {
    int i = blockIdx.x * blockDim.x + threadIdx.x;
    if (i >= NN * D) return;
    h[i] = fmaxf(h[i] + b[i % D], 0.f);
}

#define COLSUM_ROWS 256
__global__ void k_colsum(const float* __restrict__ h, float* __restrict__ g) {
    int c = threadIdx.x;                  // 128 threads
    int r0 = blockIdx.x * COLSUM_ROWS;
    float s = 0.f;
    for (int r = 0; r < COLSUM_ROWS; ++r) {
        int row = r0 + r;
        if (row < NN) s += h[(size_t)row * HIDC + c];
    }
    atomicAdd(&g[c], s);
}

__global__ void k_value(const float* __restrict__ g, const float* __restrict__ Wv1,
                        const float* __restrict__ bv1, const float* __restrict__ Wv2,
                        const float* __restrict__ bv2, float* __restrict__ out_v) {
    __shared__ float gm[HIDC];
    __shared__ float hv[HEDIM];
    int t = threadIdx.x;                  // 128 threads
    gm[t] = g[t] * (1.0f / NN);
    __syncthreads();
    if (t < HEDIM) {
        float s = bv1[t];
        for (int c = 0; c < HIDC; ++c) s += gm[c] * Wv1[t * HIDC + c];
        hv[t] = fmaxf(s, 0.f);
    }
    __syncthreads();
    if (t == 0) {
        float s = bv2[0];
        for (int j = 0; j < HEDIM; ++j) s += hv[j] * Wv2[j];
        out_v[0] = tanhf(s);
    }
}

extern "C" void kernel_launch(void* const* d_in, const int* in_sizes, int n_in,
                              void* d_out, int out_size, void* d_ws, size_t ws_size,
                              hipStream_t stream) {
    const float* x    = (const float*)d_in[0];
    const int*   ntyp = (const int*)  d_in[1];
    const int*   ei   = (const int*)  d_in[2];
    const float* ea   = (const float*)d_in[3];
    const float* emb  = (const float*)d_in[4];
    const float* We1  = (const float*)d_in[5];
    const float* be1  = (const float*)d_in[6];
    const float* We2  = (const float*)d_in[7];
    const float* be2  = (const float*)d_in[8];
    const float* Wn   = (const float*)d_in[9];
    const float* bn   = (const float*)d_in[10];
    const float* c1W  = (const float*)d_in[11];
    const float* c1as = (const float*)d_in[12];
    const float* c1ad = (const float*)d_in[13];
    const float* c1ae = (const float*)d_in[14];
    const float* c1We = (const float*)d_in[15];
    const float* c1b  = (const float*)d_in[16];
    const float* c2W  = (const float*)d_in[17];
    const float* c2as = (const float*)d_in[18];
    const float* c2ad = (const float*)d_in[19];
    const float* c2ae = (const float*)d_in[20];
    const float* c2We = (const float*)d_in[21];
    const float* c2b  = (const float*)d_in[22];
    const float* Wv1  = (const float*)d_in[23];
    const float* bv1  = (const float*)d_in[24];
    const float* Wv2  = (const float*)d_in[25];
    const float* bv2  = (const float*)d_in[26];

    float* out = (float*)d_out;           // [NN*128] final h, then [NN*128] = v
    float* ws  = (float*)d_ws;

    // ---- workspace layout (floats), with region reuse ----
    size_t o_h0   = 0;                               // NN*128
    size_t o_e    = o_h0   + (size_t)NN * 128;       // NE*64  (final edge feats)
    size_t o_tmp  = o_e    + (size_t)NE * 64;        // NE*64  (e1, then xp1 = NN*512)
    size_t o_hin  = o_tmp  + (size_t)NN * 512;       // NN*128 (inside tmp, dead before e1)
    size_t o_out1 = o_tmp  + (size_t)NE * 64;        // NN*512 (out1 -> h1 in place)
    size_t o_xp2  = o_out1 + (size_t)NN * 512;       // NN*128
    size_t o_esum = o_xp2  + (size_t)NN * 128;       // NN*64 -> loop_attr in place
    size_t o_deg  = o_esum + (size_t)NN * 64;        // NN
    size_t o_lg1  = o_deg  + (size_t)NN;             // NE*4
    size_t o_as1  = o_lg1  + (size_t)NE * 4;         // NN*4
    size_t o_ad1  = o_as1  + (size_t)NN * 4;
    size_t o_lsl1 = o_ad1  + (size_t)NN * 4;
    size_t o_m1   = o_lsl1 + (size_t)NN * 4;
    size_t o_dn1  = o_m1   + (size_t)NN * 4;
    size_t o_M1   = o_dn1  + (size_t)NN * 4;         // 4*64
    size_t o_lg2  = o_M1   + 256;                    // NE
    size_t o_as2  = o_lg2  + (size_t)NE;             // NN
    size_t o_ad2  = o_as2  + (size_t)NN;
    size_t o_lsl2 = o_ad2  + (size_t)NN;
    size_t o_m2   = o_lsl2 + (size_t)NN;
    size_t o_dn2  = o_m2   + (size_t)NN;
    size_t o_M2   = o_dn2  + (size_t)NN;             // 64
    size_t o_g    = o_M2   + 64;                     // 128

    auto nb = [](long long n) { return dim3((unsigned)((n + 255) / 256)); };
    auto gemm = [&](const float* A, const float* W, const float* b, float* C,
                    int M, int K, int Nout, int relu) {
        long long waves = (long long)(M / 16) * (Nout / 64);
        dim3 g((unsigned)((waves + 7) / 8)), bd(256);
        switch (K) {
        case 16:  wmma_gemm<16>  <<<g, bd, 0, stream>>>(A, W, b, C, M, Nout, relu); break;
        case 64:  wmma_gemm<64>  <<<g, bd, 0, stream>>>(A, W, b, C, M, Nout, relu); break;
        case 128: wmma_gemm<128> <<<g, bd, 0, stream>>>(A, W, b, C, M, Nout, relu); break;
        case 512: wmma_gemm<512> <<<g, bd, 0, stream>>>(A, W, b, C, M, Nout, relu); break;
        default: break;
        }
    };

    // zero accumulators
    k_zero<<<nb(NN), 256, 0, stream>>>(ws + o_deg, NN);
    k_zero<<<nb((long long)NN * 64), 256, 0, stream>>>(ws + o_esum, NN * 64);
    k_zero<<<nb(128), 256, 0, stream>>>(ws + o_g, 128);

    // node embed: hin = concat(x, emb[type]) ; h0 = relu(hin @ Wn^T + bn)
    k_concat<<<nb((long long)NN * 128), 256, 0, stream>>>(x, ntyp, emb, ws + o_hin);
    gemm(ws + o_hin, Wn, bn, ws + o_h0, NN, 128, 128, 1);

    // edge MLP: e1 = relu(ea @ We1^T + be1) ; e = relu(e1 @ We2^T + be2)
    gemm(ea, We1, be1, ws + o_tmp, NE, 16, 64, 1);
    gemm(ws + o_tmp, We2, be2, ws + o_e, NE, 64, 64, 1);

    // self-loop edge attr = mean of incoming e
    k_deg_esum<<<nb((long long)NE * 64), 256, 0, stream>>>(ws + o_e, ei, ws + o_esum, ws + o_deg);
    k_loopattr<<<nb((long long)NN * 64), 256, 0, stream>>>(ws + o_esum, ws + o_deg);

    // ---------------- GAT layer 1 (H=4, C=128) ----------------
    gemm(ws + o_h0, c1W, nullptr, ws + o_tmp, NN, 128, 512, 0);          // xp1 (clobbers e1)
    k_attdot<<<nb((long long)NN * 4), 256, 0, stream>>>(ws + o_tmp, c1as, c1ad,
                                                        ws + o_as1, ws + o_ad1, 4, 128);
    k_foldW<<<nb(4 * 64), 256, 0, stream>>>(c1We, c1ae, ws + o_M1, 4, 128);
    k_selfloop<<<nb((long long)NN * 4), 256, 0, stream>>>(ws + o_esum, ws + o_M1,
                                                          ws + o_as1, ws + o_ad1,
                                                          ws + o_lsl1, ws + o_m1, 4);
    k_edgelogit<<<nb((long long)NE * 4), 256, 0, stream>>>(ws + o_e, ei, ws + o_M1,
                                                           ws + o_as1, ws + o_ad1,
                                                           ws + o_lg1, ws + o_m1, 4);
    k_denom_init<<<nb((long long)NN * 4), 256, 0, stream>>>(ws + o_lsl1, ws + o_m1, ws + o_dn1, 4);
    k_denom_edge<<<nb((long long)NE * 4), 256, 0, stream>>>(ws + o_lg1, ei, ws + o_m1, ws + o_dn1, 4);
    k_agg_init<<<nb((long long)NN * 128), 256, 0, stream>>>(ws + o_tmp, ws + o_lsl1,
                                                            ws + o_m1, ws + o_dn1,
                                                            ws + o_out1, 4, 128);
    k_agg_edge<<<nb((long long)NE * 128), 256, 0, stream>>>(ws + o_tmp, ei, ws + o_lg1,
                                                            ws + o_m1, ws + o_dn1,
                                                            ws + o_out1, 4, 128);
    k_bias_relu<<<nb((long long)NN * 512), 256, 0, stream>>>(ws + o_out1, c1b, 512);  // h1

    // ---------------- GAT layer 2 (H=1, C=128) ----------------
    gemm(ws + o_out1, c2W, nullptr, ws + o_xp2, NN, 512, 128, 0);        // xp2
    k_attdot<<<nb(NN), 256, 0, stream>>>(ws + o_xp2, c2as, c2ad, ws + o_as2, ws + o_ad2, 1, 128);
    k_foldW<<<nb(64), 256, 0, stream>>>(c2We, c2ae, ws + o_M2, 1, 128);
    k_selfloop<<<nb(NN), 256, 0, stream>>>(ws + o_esum, ws + o_M2, ws + o_as2, ws + o_ad2,
                                           ws + o_lsl2, ws + o_m2, 1);
    k_edgelogit<<<nb(NE), 256, 0, stream>>>(ws + o_e, ei, ws + o_M2, ws + o_as2, ws + o_ad2,
                                            ws + o_lg2, ws + o_m2, 1);
    k_denom_init<<<nb(NN), 256, 0, stream>>>(ws + o_lsl2, ws + o_m2, ws + o_dn2, 1);
    k_denom_edge<<<nb(NE), 256, 0, stream>>>(ws + o_lg2, ei, ws + o_m2, ws + o_dn2, 1);
    k_agg_init<<<nb((long long)NN * 32), 256, 0, stream>>>(ws + o_xp2, ws + o_lsl2,
                                                           ws + o_m2, ws + o_dn2, out, 1, 128);
    k_agg_edge<<<nb((long long)NE * 32), 256, 0, stream>>>(ws + o_xp2, ei, ws + o_lg2,
                                                           ws + o_m2, ws + o_dn2, out, 1, 128);
    k_bias_relu<<<nb((long long)NN * 128), 256, 0, stream>>>(out, c2b, 128);   // final h -> d_out

    // global mean pool + value head -> d_out[NN*128]
    k_colsum<<<dim3((NN + COLSUM_ROWS - 1) / COLSUM_ROWS), dim3(128), 0, stream>>>(out, ws + o_g);
    k_value<<<dim3(1), dim3(128), 0, stream>>>(ws + o_g, Wv1, bv1, Wv2, bv2, out + (size_t)NN * 128);
}
